// Decoder_9251359556330
// MI455X (gfx1250) — matrix operand order repaired and checked by reading, not compiled, not run on hardware
//
#include <hip/hip_runtime.h>
#include <hip/hip_bf16.h>
#include <math.h>

// ---------------------------------------------------------------------------
// MI455X (gfx1250) decoder: bf16 WMMA GEMMs + TDM-staged LSTM recurrence.
//   B=32 S=128 T=64 VOC=32000 E=256 U=512 EU=512, D1=E+EU=768, 4U=2048
// ---------------------------------------------------------------------------

#define BB   32
#define SS   128
#define TT_  64
#define VOC  32000
#define EE   256
#define UU   512
#define EUU  512
#define D1   768
#define G4   2048           // 4*U
#define MBT  2048           // B*T
#define MBS  4096           // B*S
#define REC_WGS 16

typedef __bf16 bf16_t;
typedef bf16_t v16bf __attribute__((ext_vector_type(16)));
typedef float  v8f   __attribute__((ext_vector_type(8)));
typedef unsigned int v4u  __attribute__((ext_vector_type(4)));
typedef int          v4i  __attribute__((ext_vector_type(4)));
typedef int          v8i  __attribute__((ext_vector_type(8)));

union FragB16 { v16bf v; v4u q[2]; };

__device__ __forceinline__ unsigned short f32_to_bf16(float f) {
  unsigned u = __float_as_uint(f);
  unsigned rnd = 0x7FFFu + ((u >> 16) & 1u);
  return (unsigned short)((u + rnd) >> 16);
}
__device__ __forceinline__ float sigm(float x) { return 1.f / (1.f + __expf(-x)); }

// ---------------------------------------------------------------------------
// Elementwise conversion / transpose kernels
// ---------------------------------------------------------------------------
__global__ void k_conv_bf16(const float* __restrict__ src,
                            unsigned short* __restrict__ dst, int n) {
  int i = blockIdx.x * blockDim.x + threadIdx.x;
  if (i < n) dst[i] = f32_to_bf16(src[i]);
}

// src [K,N] f32 row-major -> dst [N,K] bf16 row-major
__global__ void k_transpose_bf16(const float* __restrict__ src,
                                 unsigned short* __restrict__ dst, int K, int N) {
  int i = blockIdx.x * blockDim.x + threadIdx.x;
  if (i >= K * N) return;
  int k = i / N, n = i - k * N;
  dst[(size_t)n * K + k] = f32_to_bf16(src[i]);
}

__global__ void k_init_state(const float* __restrict__ h0, const float* __restrict__ c0,
                             unsigned short* __restrict__ h_cur, float* __restrict__ c_cur) {
  int i = blockIdx.x * blockDim.x + threadIdx.x;
  if (i < BB * UU) { h_cur[i] = f32_to_bf16(h0[i]); c_cur[i] = c0[i]; }
}

__global__ void k_copy_states(const float* __restrict__ hT, const float* __restrict__ cT,
                              float* __restrict__ oh, float* __restrict__ oc) {
  int i = blockIdx.x * blockDim.x + threadIdx.x;
  if (i < BB * UU) { oh[i] = hT[i]; oc[i] = cT[i]; }
}

__global__ void k_zero64(unsigned long long* p) { *p = 0ull; }

// ---------------------------------------------------------------------------
// Generic bf16 WMMA GEMM: C[M,N] = A[M,K](bf16) * BT[N,K](bf16)^T + bias[N]
// block = 256 threads = 8 waves (2 M x 4 N), wave tile = 16x64.
// 2-deep software pipeline with two STATICALLY-NAMED buffer sets (no dynamic
// indexing -> no scratch spills): loads of k-step i+1 overlap WMMAs of step i.
// Requires K % 64 == 0 (all call sites: 512 or 768).
// grid = (N/256, M/32); all call sites are exact multiples.
// ---------------------------------------------------------------------------
__global__ __launch_bounds__(256) void k_gemm_bf16(
    const unsigned short* __restrict__ A,
    const unsigned short* __restrict__ BT,
    const float* __restrict__ bias,
    float* __restrict__ C, int M, int K, int N) {
  int wave = threadIdx.x >> 5;
  int lane = threadIdx.x & 31;
  int half = lane >> 4, r = lane & 15;
  int row0 = blockIdx.y * 32 + (wave >> 2) * 16;
  int col0 = blockIdx.x * 256 + (wave & 3) * 64;
  if (row0 >= M || col0 >= N) return;

  const unsigned short* ap  = A  + (size_t)(row0 + r) * K + half * 16;
  const unsigned short* bp0 = BT + (size_t)(col0 + 0 * 16 + r) * K + half * 16;
  const unsigned short* bp1 = BT + (size_t)(col0 + 1 * 16 + r) * K + half * 16;
  const unsigned short* bp2 = BT + (size_t)(col0 + 2 * 16 + r) * K + half * 16;
  const unsigned short* bp3 = BT + (size_t)(col0 + 3 * 16 + r) * K + half * 16;

  v8f acc0, acc1, acc2, acc3;
#pragma unroll
  for (int i = 0; i < 8; ++i) { acc0[i] = 0.f; acc1[i] = 0.f; acc2[i] = 0.f; acc3[i] = 0.f; }

  FragB16 faA, fbA0, fbA1, fbA2, fbA3;   // buffer set A
  FragB16 faB, fbB0, fbB1, fbB2, fbB3;   // buffer set B

  auto ldA = [&](int kb) {
    const unsigned short* a = ap + kb;
    faA.q[0] = *(const v4u*)(a);        faA.q[1] = *(const v4u*)(a + 8);
    const unsigned short* b0 = bp0 + kb;
    fbA0.q[0] = *(const v4u*)(b0);      fbA0.q[1] = *(const v4u*)(b0 + 8);
    const unsigned short* b1 = bp1 + kb;
    fbA1.q[0] = *(const v4u*)(b1);      fbA1.q[1] = *(const v4u*)(b1 + 8);
    const unsigned short* b2 = bp2 + kb;
    fbA2.q[0] = *(const v4u*)(b2);      fbA2.q[1] = *(const v4u*)(b2 + 8);
    const unsigned short* b3 = bp3 + kb;
    fbA3.q[0] = *(const v4u*)(b3);      fbA3.q[1] = *(const v4u*)(b3 + 8);
  };
  auto ldB = [&](int kb) {
    const unsigned short* a = ap + kb;
    faB.q[0] = *(const v4u*)(a);        faB.q[1] = *(const v4u*)(a + 8);
    const unsigned short* b0 = bp0 + kb;
    fbB0.q[0] = *(const v4u*)(b0);      fbB0.q[1] = *(const v4u*)(b0 + 8);
    const unsigned short* b1 = bp1 + kb;
    fbB1.q[0] = *(const v4u*)(b1);      fbB1.q[1] = *(const v4u*)(b1 + 8);
    const unsigned short* b2 = bp2 + kb;
    fbB2.q[0] = *(const v4u*)(b2);      fbB2.q[1] = *(const v4u*)(b2 + 8);
    const unsigned short* b3 = bp3 + kb;
    fbB3.q[0] = *(const v4u*)(b3);      fbB3.q[1] = *(const v4u*)(b3 + 8);
  };
  auto mmA = [&]() {
    acc0 = __builtin_amdgcn_wmma_f32_16x16x32_bf16(false, faA.v, false, fbA0.v, (short)0, acc0, false, false);
    acc1 = __builtin_amdgcn_wmma_f32_16x16x32_bf16(false, faA.v, false, fbA1.v, (short)0, acc1, false, false);
    acc2 = __builtin_amdgcn_wmma_f32_16x16x32_bf16(false, faA.v, false, fbA2.v, (short)0, acc2, false, false);
    acc3 = __builtin_amdgcn_wmma_f32_16x16x32_bf16(false, faA.v, false, fbA3.v, (short)0, acc3, false, false);
  };
  auto mmB = [&]() {
    acc0 = __builtin_amdgcn_wmma_f32_16x16x32_bf16(false, faB.v, false, fbB0.v, (short)0, acc0, false, false);
    acc1 = __builtin_amdgcn_wmma_f32_16x16x32_bf16(false, faB.v, false, fbB1.v, (short)0, acc1, false, false);
    acc2 = __builtin_amdgcn_wmma_f32_16x16x32_bf16(false, faB.v, false, fbB2.v, (short)0, acc2, false, false);
    acc3 = __builtin_amdgcn_wmma_f32_16x16x32_bf16(false, faB.v, false, fbB3.v, (short)0, acc3, false, false);
  };

  ldA(0);
  ldB(32);
  int kb = 0;
  for (; kb + 64 < K; kb += 64) {
    mmA(); ldA(kb + 64);
    mmB(); ldB(kb + 96);
  }
  mmA();
  mmB();

  {
    int col = col0 + r;
    float bv0 = bias[col];
    float bv1 = bias[col + 16];
    float bv2 = bias[col + 32];
    float bv3 = bias[col + 48];
#pragma unroll
    for (int i = 0; i < 8; ++i) {
      size_t row = (size_t)(row0 + half * 8 + i) * N + col;
      C[row]      = acc0[i] + bv0;
      C[row + 16] = acc1[i] + bv1;
      C[row + 32] = acc2[i] + bv2;
      C[row + 48] = acc3[i] + bv3;
    }
  }
}

// ---------------------------------------------------------------------------
// Attention: hW2[b,u] = estado_h[b]@W2 + b2
// ---------------------------------------------------------------------------
__global__ void k_hW2(const float* __restrict__ h0, const float* __restrict__ W2,
                      const float* __restrict__ b2, float* __restrict__ hW2) {
  int idx = blockIdx.x * blockDim.x + threadIdx.x;
  if (idx >= BB * UU) return;
  int b = idx >> 9, u = idx & 511;
  float s = b2[u];
  for (int k = 0; k < UU; ++k) s += h0[b * UU + k] * W2[k * UU + u];
  hW2[idx] = s;
}

// score[b,s] = Va . tanh(TT[row] + hW2[b]) + bV ; one wave per row
__global__ __launch_bounds__(256) void k_attn_score(
    const float* __restrict__ TT, const float* __restrict__ hW2,
    const float* __restrict__ Va, const float* __restrict__ bV,
    float* __restrict__ score) {
  int row = blockIdx.x * 8 + (threadIdx.x >> 5);
  int lane = threadIdx.x & 31;
  int b = row >> 7;
  float sum = 0.f;
  for (int u = lane; u < UU; u += 32)
    sum += tanhf(TT[(size_t)row * UU + u] + hW2[b * UU + u]) * Va[u];
  for (int off = 16; off > 0; off >>= 1) sum += __shfl_down(sum, off);
  if (lane == 0) score[row] = sum + bV[0];
}

// softmax over S, pesos -> d_out, ctx = sum_s w * enc
__global__ __launch_bounds__(256) void k_softmax_ctx(
    const float* __restrict__ score, const float* __restrict__ enc,
    float* __restrict__ pesos, float* __restrict__ ctx_f32,
    unsigned short* __restrict__ ctx_bf16) {
  int b = blockIdx.x, tid = threadIdx.x;
  __shared__ float w[SS];
  __shared__ float red[256];
  float v = (tid < SS) ? score[b * SS + tid] : -1e30f;
  red[tid] = v;
  __syncthreads();
  for (int s = 128; s > 0; s >>= 1) {
    if (tid < s) red[tid] = fmaxf(red[tid], red[tid + s]);
    __syncthreads();
  }
  float mx = red[0];
  __syncthreads();
  float e = (tid < SS) ? __expf(v - mx) : 0.f;
  red[tid] = e;
  __syncthreads();
  for (int s = 128; s > 0; s >>= 1) {
    if (tid < s) red[tid] += red[tid + s];
    __syncthreads();
  }
  float denom = red[0];
  if (tid < SS) { float wv = e / denom; w[tid] = wv; pesos[b * SS + tid] = wv; }
  __syncthreads();
  for (int u = tid; u < EUU; u += 256) {
    float s = 0.f;
    for (int k = 0; k < SS; ++k) s += w[k] * enc[((size_t)b * SS + k) * EUU + u];
    ctx_f32[b * EUU + u] = s;
    ctx_bf16[b * EUU + u] = f32_to_bf16(s);
  }
}

// xcat[b*T+t, 0:256]=emb[x], [256:768]=ctx[b]  (bf16)
__global__ void k_xcat(const int* __restrict__ x, const float* __restrict__ emb,
                       const unsigned short* __restrict__ ctx_bf16,
                       unsigned short* __restrict__ xcat) {
  int idx = blockIdx.x * blockDim.x + threadIdx.x;
  if (idx >= MBT * D1) return;
  int row = idx / D1, col = idx - row * D1;
  int b = row >> 6;  // T=64
  if (col < EE) {
    int tok = x[row];
    xcat[idx] = f32_to_bf16(emb[(size_t)tok * EE + col]);
  } else {
    xcat[idx] = ctx_bf16[b * EUU + (col - EE)];
  }
}

// ---------------------------------------------------------------------------
// TDM: load h_cur [32 x 512] bf16 tile into LDS (2D descriptor, data_size=2B)
// 6-arg toolchain form: (g0 v4u, g1 v8i, g2 v4i, g3 v4i, g4 v8i, cpol)
// ---------------------------------------------------------------------------
__device__ __forceinline__ void tdm_load_h(unsigned short* lds_dst,
                                           const unsigned short* gsrc) {
  unsigned long long ga = (unsigned long long)(size_t)gsrc;
  unsigned lds_off = (unsigned)(size_t)lds_dst;  // LDS aperture: low 32 bits
  v4u g0;
  g0[0] = 1u;                                       // count=1, user mode
  g0[1] = lds_off;                                  // lds_addr [63:32]
  g0[2] = (unsigned)(ga & 0xFFFFFFFFu);             // global_addr lo
  g0[3] = (unsigned)((ga >> 32) & 0x01FFFFFFu) | (2u << 30);  // addr hi | type=2
  v8i g1;
  g1[0] = (1 << 16);                  // data_size=1 (2 bytes), no mask/flags
  g1[1] = (UU & 0xFFFF) << 16;        // tensor_dim0 lo -> bits[63:48]
  g1[2] = (UU >> 16) | ((BB & 0xFFFF) << 16);  // dim0 hi | tensor_dim1 lo
  g1[3] = (BB >> 16) | ((UU & 0xFFFF) << 16);  // dim1 hi | tile_dim0=512
  g1[4] = BB;                         // tile_dim1=32, tile_dim2=0
  g1[5] = UU;                         // tensor_dim0_stride lo = 512
  g1[6] = 0;
  g1[7] = 0;
  v4i g2; g2[0] = 0; g2[1] = 0; g2[2] = 0; g2[3] = 0;
  v4i g3; g3[0] = 0; g3[1] = 0; g3[2] = 0; g3[3] = 0;
  v8i g4;
#pragma unroll
  for (int i = 0; i < 8; ++i) g4[i] = 0;
  __builtin_amdgcn_tensor_load_to_lds(g0, g1, g2, g3, g4, 0);
}

// ---------------------------------------------------------------------------
// LSTM recurrence: 16 persistent WGs, WG j owns hidden units [j*32, j*32+32).
// Per step: z = Xp[:,t,:] (precomputed x@Wx+b) + h@Wh via WMMA, gates, update.
// h staged to LDS each step with the Tensor Data Mover; steps separated by a
// release/acquire global-atomic grid barrier. K-loop uses two statically-named
// pipeline buffer sets (no dynamic indexing -> no scratch spills).
// ---------------------------------------------------------------------------
__global__ __launch_bounds__(256) void k_lstm_rec(
    const float* __restrict__ Xp,            // [B*T, 2048] f32
    const unsigned short* __restrict__ WhT,  // [2048, 512] bf16
    unsigned short* __restrict__ h_cur,      // [B,512] bf16 in/out
    float* __restrict__ c_cur,               // [B,512] f32 in/out (final)
    unsigned short* __restrict__ h_all,      // [B*T,512] bf16 out
    float* __restrict__ hT_f32,              // [B,512] f32 final h
    unsigned long long* __restrict__ bar) {
  const int j = blockIdx.x;           // 0..15
  const int tid = threadIdx.x;
  const int wave = tid >> 5, lane = tid & 31, half = lane >> 4, r = lane & 15;

  __shared__ unsigned short h_lds[BB * UU];  // 32 KB
  __shared__ float zbuf[BB * 128];           // 16 KB
  __shared__ float c_lds[BB * 32];           // 4 KB

  for (int i = tid; i < BB * 32; i += 256) {
    int b = i >> 5, cc = i & 31;
    c_lds[i] = c_cur[b * UU + j * 32 + cc];
  }
  __syncthreads();

  unsigned long long epoch = 0;
  const int lc = wave * 16 + r;               // local z column 0..127
  const int g = lc >> 5, cc = lc & 31;        // gate, unit-within-slice
  const int gcol = g * UU + j * 32 + cc;      // column of Wh / Xp
  const unsigned short* whrow = WhT + (size_t)gcol * UU + half * 16;
  const unsigned short* a0row = &h_lds[(0 + r) * UU + half * 16];
  const unsigned short* a1row = &h_lds[(16 + r) * UU + half * 16];

  for (int t = 0; t < TT_; ++t) {
    // ---- stage h[32,512] bf16 into LDS via TDM ----
    if (tid == 0) {
      tdm_load_h(h_lds, h_cur);
      __builtin_amdgcn_s_wait_tensorcnt(0);
    }
    __syncthreads();

    // ---- z tile: two 16x16 WMMA accumulators (b 0-15, b 16-31) ----
    v8f acc0, acc1;
#pragma unroll
    for (int i = 0; i < 8; ++i) { acc0[i] = 0.f; acc1[i] = 0.f; }

    FragB16 fa0A, fa1A, fbA;    // buffer set A
    FragB16 fa0B, fa1B, fbB;    // buffer set B
    auto ldA = [&](int kb) {
      const unsigned short* a0 = a0row + kb;
      fa0A.q[0] = *(const v4u*)(a0);  fa0A.q[1] = *(const v4u*)(a0 + 8);
      const unsigned short* a1 = a1row + kb;
      fa1A.q[0] = *(const v4u*)(a1);  fa1A.q[1] = *(const v4u*)(a1 + 8);
      const unsigned short* bp = whrow + kb;
      fbA.q[0] = *(const v4u*)(bp);   fbA.q[1] = *(const v4u*)(bp + 8);
    };
    auto ldB = [&](int kb) {
      const unsigned short* a0 = a0row + kb;
      fa0B.q[0] = *(const v4u*)(a0);  fa0B.q[1] = *(const v4u*)(a0 + 8);
      const unsigned short* a1 = a1row + kb;
      fa1B.q[0] = *(const v4u*)(a1);  fa1B.q[1] = *(const v4u*)(a1 + 8);
      const unsigned short* bp = whrow + kb;
      fbB.q[0] = *(const v4u*)(bp);   fbB.q[1] = *(const v4u*)(bp + 8);
    };
    auto mmA = [&]() {
      acc0 = __builtin_amdgcn_wmma_f32_16x16x32_bf16(false, fa0A.v, false, fbA.v, (short)0, acc0, false, false);
      acc1 = __builtin_amdgcn_wmma_f32_16x16x32_bf16(false, fa1A.v, false, fbA.v, (short)0, acc1, false, false);
    };
    auto mmB = [&]() {
      acc0 = __builtin_amdgcn_wmma_f32_16x16x32_bf16(false, fa0B.v, false, fbB.v, (short)0, acc0, false, false);
      acc1 = __builtin_amdgcn_wmma_f32_16x16x32_bf16(false, fa1B.v, false, fbB.v, (short)0, acc1, false, false);
    };

    ldA(0);
    ldB(32);
    int kb = 0;
    for (; kb + 64 < UU; kb += 64) {
      mmA(); ldA(kb + 64);
      mmB(); ldB(kb + 96);
    }
    mmA();
    mmB();

#pragma unroll
    for (int i = 0; i < 8; ++i) {
      int b0 = half * 8 + i;
      float xp0 = Xp[((size_t)(b0 * TT_ + t)) * G4 + gcol];
      zbuf[b0 * 128 + lc] = acc0[i] + xp0;
      int b1 = 16 + half * 8 + i;
      float xp1 = Xp[((size_t)(b1 * TT_ + t)) * G4 + gcol];
      zbuf[b1 * 128 + lc] = acc1[i] + xp1;
    }
    __syncthreads();

    // ---- gates + state update for owned 32x32 (b, unit) block ----
    for (int i = tid; i < BB * 32; i += 256) {
      int b = i >> 5, u = i & 31;
      float zi = zbuf[b * 128 + 0 * 32 + u];
      float zf = zbuf[b * 128 + 1 * 32 + u];
      float zg = zbuf[b * 128 + 2 * 32 + u];
      float zo = zbuf[b * 128 + 3 * 32 + u];
      float cold = c_lds[i];
      float cnew = sigm(zf) * cold + sigm(zi) * tanhf(zg);
      float hnew = sigm(zo) * tanhf(cnew);
      c_lds[i] = cnew;
      int gidx = b * UU + j * 32 + u;
      unsigned short hb = f32_to_bf16(hnew);
      h_all[((size_t)(b * TT_ + t)) * UU + j * 32 + u] = hb;
      h_cur[gidx] = hb;
      if (t == TT_ - 1) { hT_f32[gidx] = hnew; c_cur[gidx] = cnew; }
    }
    __threadfence();
    __syncthreads();

    // ---- grid barrier across the 16 WGs ----
    if (tid == 0) {
      unsigned long long need = (epoch + 1) * (unsigned long long)REC_WGS;
      atomicAdd(bar, 1ull);
      while (__hip_atomic_load(bar, __ATOMIC_ACQUIRE, __HIP_MEMORY_SCOPE_AGENT) < need)
        __builtin_amdgcn_s_sleep(1);
    }
    epoch++;
    __syncthreads();
  }
}

// ---------------------------------------------------------------------------
// launcher
// ---------------------------------------------------------------------------
extern "C" void kernel_launch(void* const* d_in, const int* in_sizes, int n_in,
                              void* d_out, int out_size, void* d_ws, size_t ws_size,
                              hipStream_t stream) {
  const int*   x    = (const int*)d_in[0];
  const float* enc  = (const float*)d_in[1];
  const float* h0   = (const float*)d_in[2];
  const float* c0   = (const float*)d_in[3];
  const float* emb  = (const float*)d_in[4];
  const float* W1   = (const float*)d_in[5];
  const float* b1   = (const float*)d_in[6];
  const float* W2   = (const float*)d_in[7];
  const float* b2   = (const float*)d_in[8];
  const float* Va   = (const float*)d_in[9];
  const float* bV   = (const float*)d_in[10];
  const float* Wx1  = (const float*)d_in[11];
  const float* Wh1  = (const float*)d_in[12];
  const float* bl1  = (const float*)d_in[13];
  const float* Wx2  = (const float*)d_in[14];
  const float* Wh2  = (const float*)d_in[15];
  const float* bl2  = (const float*)d_in[16];
  const float* Wfc  = (const float*)d_in[17];
  const float* bfc  = (const float*)d_in[18];

  // workspace carve-up (256B aligned)
  char* base = (char*)d_ws;
  size_t off = 0;
  auto alloc = [&](size_t bytes) -> char* {
    char* p = base + off;
    off = (off + bytes + 255) & ~(size_t)255;
    return p;
  };
  unsigned short* enc_bf  = (unsigned short*)alloc((size_t)MBS * EUU * 2);
  unsigned short* W1T     = (unsigned short*)alloc((size_t)UU * EUU * 2);
  unsigned short* Wx1T    = (unsigned short*)alloc((size_t)G4 * D1 * 2);
  unsigned short* Wh1T    = (unsigned short*)alloc((size_t)G4 * UU * 2);
  unsigned short* Wx2T    = (unsigned short*)alloc((size_t)G4 * UU * 2);
  unsigned short* Wh2T    = (unsigned short*)alloc((size_t)G4 * UU * 2);
  unsigned short* WfcT    = (unsigned short*)alloc((size_t)VOC * UU * 2);
  float*          hW2     = (float*)alloc((size_t)BB * UU * 4);
  float*          TTb     = (float*)alloc((size_t)MBS * UU * 4);
  float*          score   = (float*)alloc((size_t)BB * SS * 4);
  float*          ctx_f32 = (float*)alloc((size_t)BB * EUU * 4);
  unsigned short* ctx_bf  = (unsigned short*)alloc((size_t)BB * EUU * 2);
  unsigned short* xcat    = (unsigned short*)alloc((size_t)MBT * D1 * 2);
  float*          Xp1     = (float*)alloc((size_t)MBT * G4 * 4);
  float*          Xp2     = (float*)alloc((size_t)MBT * G4 * 4);
  unsigned short* h1_all  = (unsigned short*)alloc((size_t)MBT * UU * 2);
  unsigned short* h2_all  = (unsigned short*)alloc((size_t)MBT * UU * 2);
  unsigned short* h_cur   = (unsigned short*)alloc((size_t)BB * UU * 2);
  float*          c_cur   = (float*)alloc((size_t)BB * UU * 4);
  float*          hT_f32  = (float*)alloc((size_t)BB * UU * 4);
  unsigned long long* bar = (unsigned long long*)alloc(256);

  float* out_sal = (float*)d_out;
  float* out_hT  = out_sal + (size_t)BB * TT_ * VOC;
  float* out_cT  = out_hT + BB * UU;
  float* out_pes = out_cT + BB * UU;

  auto gblk = [](int n) { return dim3((unsigned)((n + 255) / 256)); };

  // 1) weight conversion / transposition to bf16 [N,K]
  k_conv_bf16<<<gblk(MBS * EUU), 256, 0, stream>>>(enc, enc_bf, MBS * EUU);
  k_transpose_bf16<<<gblk(EUU * UU), 256, 0, stream>>>(W1, W1T, EUU, UU);
  k_transpose_bf16<<<gblk(D1 * G4), 256, 0, stream>>>(Wx1, Wx1T, D1, G4);
  k_transpose_bf16<<<gblk(UU * G4), 256, 0, stream>>>(Wh1, Wh1T, UU, G4);
  k_transpose_bf16<<<gblk(UU * G4), 256, 0, stream>>>(Wx2, Wx2T, UU, G4);
  k_transpose_bf16<<<gblk(UU * G4), 256, 0, stream>>>(Wh2, Wh2T, UU, G4);
  k_transpose_bf16<<<gblk(UU * VOC), 256, 0, stream>>>(Wfc, WfcT, UU, VOC);

  // 2) attention
  k_hW2<<<gblk(BB * UU), 256, 0, stream>>>(h0, W2, b2, hW2);
  k_gemm_bf16<<<dim3(UU / 256, MBS / 32), 256, 0, stream>>>(enc_bf, W1T, b1, TTb,
                                                            MBS, EUU, UU);
  k_attn_score<<<dim3(MBS / 8), 256, 0, stream>>>(TTb, hW2, Va, bV, score);
  k_softmax_ctx<<<dim3(BB), 256, 0, stream>>>(score, enc, out_pes, ctx_f32, ctx_bf);

  // 3) embed + concat, LSTM input precompute (layer 1)
  k_init_state<<<gblk(BB * UU), 256, 0, stream>>>(h0, c0, h_cur, c_cur);
  k_xcat<<<gblk(MBT * D1), 256, 0, stream>>>(x, emb, ctx_bf, xcat);
  k_gemm_bf16<<<dim3(G4 / 256, MBT / 32), 256, 0, stream>>>(xcat, Wx1T, bl1, Xp1,
                                                            MBT, D1, G4);

  // 4) LSTM layer 1 recurrence
  k_zero64<<<1, 1, 0, stream>>>(bar);
  k_lstm_rec<<<dim3(REC_WGS), 256, 0, stream>>>(Xp1, Wh1T, h_cur, c_cur, h1_all,
                                                hT_f32, bar);

  // 5) layer 2 input precompute + recurrence (init state = layer1 final)
  k_gemm_bf16<<<dim3(G4 / 256, MBT / 32), 256, 0, stream>>>(h1_all, Wx2T, bl2, Xp2,
                                                            MBT, UU, G4);
  k_zero64<<<1, 1, 0, stream>>>(bar);
  k_lstm_rec<<<dim3(REC_WGS), 256, 0, stream>>>(Xp2, Wh2T, h_cur, c_cur, h2_all,
                                                hT_f32, bar);

  // 6) output projection (dominant GEMM) straight into d_out
  k_gemm_bf16<<<dim3(VOC / 256, MBT / 32), 256, 0, stream>>>(h2_all, WfcT, bfc,
                                                             out_sal, MBT, UU, VOC);
  // 7) final states
  k_copy_states<<<gblk(BB * UU), 256, 0, stream>>>(hT_f32, c_cur, out_hT, out_cT);
}
